// GNNStack_17635135717620
// MI455X (gfx1250) — matrix operand order repaired and checked by reading, not compiled, not run on hardware
//
#include <hip/hip_runtime.h>

typedef __attribute__((ext_vector_type(16))) _Float16 v16h;
typedef __attribute__((ext_vector_type(8)))  _Float16 v8h;
typedef __attribute__((ext_vector_type(8)))  float    v8f;

#define N_NODES 50000
#define E_EDGES 800000
#define N_EVAL  100000
#define DH      128      // D == H == 128
#define EPS     1e-12f

// ---------------------------------------------------------------------------
// Fragment helpers.  CDNA5 V_WMMA_F32_16X16X32_F16 layouts (ISA 7.12.2):
//   A (16x32 MxK): lane l -> M = l%16; element i -> K = i + 8*(i>=8) + 8*(l>=16)
//   B (32x16 KxN): lane l -> N = l%16; element i -> K = i + 8*(i>=8) + 8*(l>=16)
//   C (16x16):     lane l, vgpr v -> M = v + 8*(l>=16), N = l%16
// ---------------------------------------------------------------------------

// A stays row-major As[64][32]; its two 8-element K-runs are 16B aligned.
__device__ __forceinline__ v16h frag_a(const _Float16* __restrict__ As,
                                       int mBase, int lane) {
  const _Float16* p = As + (mBase + (lane & 15)) * 32 + ((lane >> 4) << 3);
  v8h lo = *(const v8h*)p;          // K = koff .. koff+7
  v8h hi = *(const v8h*)(p + 16);   // K = koff+16 .. koff+23
  return __builtin_shufflevector(lo, hi, 0, 1, 2, 3, 4, 5, 6, 7,
                                 8, 9, 10, 11, 12, 13, 14, 15);
}

// B is staged fragment-major: Bs[((ct*32 + lane) << 4) + i]  (8 KB for 32x128)
__device__ __forceinline__ v16h frag_b(const _Float16* __restrict__ Bs,
                                       int ct, int lane) {
  return *(const v16h*)(Bs + ((ct * 32 + lane) << 4));
}

// Stage one 32x128 weight chunk (rows kc*32..kc*32+31 of W[ldK x 128]) into
// fragment-major order.  Coalesced float4 reads; 4 x b16 LDS stores each.
__device__ __forceinline__ void stage_b(const float* __restrict__ W, int kc,
                                        _Float16* __restrict__ Bs, int tid) {
#pragma unroll
  for (int j = 0; j < 4; ++j) {
    int e = j * 1024 + tid * 4;          // 4096 floats total
    int r = e >> 7, c = e & 127;         // r: K within chunk, c: N
    const float4 w4 = *(const float4*)(W + (size_t)(kc * 32 + r) * DH + c);
    int base = (((c >> 4) * 32 + (((r >> 3) & 1) << 4) + (c & 15)) << 4) + ((r & 7) + ((r >> 4) << 3));
    Bs[base]      = (_Float16)w4.x;
    Bs[base + 16] = (_Float16)w4.y;
    Bs[base + 32] = (_Float16)w4.z;
    Bs[base + 48] = (_Float16)w4.w;
  }
}

__device__ __forceinline__ void cvt_store8(_Float16* dst, float4 a, float4 b) {
  v8h h;
  h[0] = (_Float16)a.x; h[1] = (_Float16)a.y;
  h[2] = (_Float16)a.z; h[3] = (_Float16)a.w;
  h[4] = (_Float16)b.x; h[5] = (_Float16)b.y;
  h[6] = (_Float16)b.z; h[7] = (_Float16)b.w;
  *(v8h*)dst = h;
}

__device__ __forceinline__ v8f wmma_f16(v16h a, v16h b, v8f c) {
  return __builtin_amdgcn_wmma_f32_16x16x32_f16(false, a, false, b, (short)0, c,
                                                false, false);
}

// ---------------- Kernel 1: H = relu(X @ W + b), K = 128 ----------------
__global__ __launch_bounds__(256) void lin_relu_kernel(
    const float* __restrict__ X, const float* __restrict__ W,
    const float* __restrict__ bias, float* __restrict__ Hout, int nrows) {
  __shared__ __align__(32) _Float16 As[64 * 32];
  __shared__ __align__(32) _Float16 Bs[8 * 32 * 16];
  const int tid = threadIdx.x, lane = tid & 31, wave = tid >> 5;
  const int rowBase = blockIdx.x * 64;
  const int rt = wave & 3, ct0 = (wave >> 2) * 4;

  v8f acc[4];
#pragma unroll
  for (int t = 0; t < 4; ++t)
#pragma unroll
    for (int v = 0; v < 8; ++v) acc[t][v] = 0.0f;

  const int ar = tid >> 2, ac = (tid & 3) * 8;   // A staging: 8 elems/thread
  for (int kc = 0; kc < DH / 32; ++kc) {
    {
      int grow = rowBase + ar;
      float4 x0{0, 0, 0, 0}, x1{0, 0, 0, 0};
      if (grow < nrows) {
        const float* p = X + (size_t)grow * DH + kc * 32 + ac;
        x0 = *(const float4*)p;
        x1 = *(const float4*)(p + 4);
      }
      cvt_store8(As + ar * 32 + ac, x0, x1);
    }
    stage_b(W, kc, Bs, tid);
    __syncthreads();
    v16h a = frag_a(As, rt * 16, lane);
#pragma unroll
    for (int t = 0; t < 4; ++t)
      acc[t] = wmma_f16(a, frag_b(Bs, ct0 + t, lane), acc[t]);
    __syncthreads();
  }
#pragma unroll
  for (int t = 0; t < 4; ++t) {
    int n = (ct0 + t) * 16 + (lane & 15);
    float bv = bias[n];
#pragma unroll
    for (int v = 0; v < 8; ++v) {
      int grow = rowBase + rt * 16 + v + ((lane >> 4) << 3);
      if (grow < nrows) {
        float x = acc[t][v] + bv;
        Hout[(size_t)grow * DH + n] = x > 0.0f ? x : 0.0f;
      }
    }
  }
}

// ---------------- Kernel 2: zero fill ----------------
__global__ __launch_bounds__(256) void zero_kernel(float4* __restrict__ p, int n4) {
  int i = blockIdx.x * 256 + threadIdx.x;
  if (i < n4) p[i] = float4{0.f, 0.f, 0.f, 0.f};
}

// ---------------- Kernel 3: agg[dst] += h[src], one wave per edge ----------------
__global__ __launch_bounds__(256) void scatter_add_kernel(
    const float* __restrict__ Hsrc, const int* __restrict__ eidx,
    float* __restrict__ AGG, int nedges) {
  int edge = blockIdx.x * 8 + (threadIdx.x >> 5);
  if (edge >= nedges) return;
  int lane = threadIdx.x & 31;
  int s = eidx[edge];             // edge_index[0][e]  (source: message h[src])
  int d = eidx[nedges + edge];    // edge_index[1][e]  (dest:   segment_sum)
  const float4 v = *(const float4*)(Hsrc + (size_t)s * DH + lane * 4);
  float* dst = AGG + (size_t)d * DH + lane * 4;
  atomicAdd(dst + 0, v.x);
  atomicAdd(dst + 1, v.y);
  atomicAdd(dst + 2, v.z);
  atomicAdd(dst + 3, v.w);
}

// ---- Kernel 4: Z = l2norm_rows(relu([AGG | XIN] @ W + b)), K = 256 ----
__global__ __launch_bounds__(256) void agg_norm_kernel(
    const float* __restrict__ AGG, const float* __restrict__ XIN,
    const float* __restrict__ W, const float* __restrict__ bias,
    float* __restrict__ OUT, int nrows) {
  __shared__ __align__(32) _Float16 As[64 * 32];
  __shared__ __align__(32) _Float16 Bs[8 * 32 * 16];
  __shared__ float Zs[64 * DH];
  __shared__ float nrm[64];
  const int tid = threadIdx.x, lane = tid & 31, wave = tid >> 5;
  const int rowBase = blockIdx.x * 64;
  const int rt = wave & 3, ct0 = (wave >> 2) * 4;

  v8f acc[4];
#pragma unroll
  for (int t = 0; t < 4; ++t)
#pragma unroll
    for (int v = 0; v < 8; ++v) acc[t][v] = 0.0f;

  const int ar = tid >> 2, ac = (tid & 3) * 8;
  for (int kc = 0; kc < (2 * DH) / 32; ++kc) {
    {
      int grow = rowBase + ar;
      int col = kc * 32 + ac;                       // 8-run never crosses DH
      float4 x0{0, 0, 0, 0}, x1{0, 0, 0, 0};
      if (grow < nrows) {
        const float* src = (col < DH) ? (AGG + (size_t)grow * DH + col)
                                      : (XIN + (size_t)grow * DH + (col - DH));
        x0 = *(const float4*)src;
        x1 = *(const float4*)(src + 4);
      }
      cvt_store8(As + ar * 32 + ac, x0, x1);
    }
    stage_b(W, kc, Bs, tid);
    __syncthreads();
    v16h a = frag_a(As, rt * 16, lane);
#pragma unroll
    for (int t = 0; t < 4; ++t)
      acc[t] = wmma_f16(a, frag_b(Bs, ct0 + t, lane), acc[t]);
    __syncthreads();
  }
  // bias + relu into LDS tile
#pragma unroll
  for (int t = 0; t < 4; ++t) {
    int n = (ct0 + t) * 16 + (lane & 15);
    float bv = bias[n];
#pragma unroll
    for (int v = 0; v < 8; ++v) {
      int m = rt * 16 + v + ((lane >> 4) << 3);
      float x = acc[t][v] + bv;
      Zs[m * DH + n] = x > 0.0f ? x : 0.0f;
    }
  }
  __syncthreads();
  if (tid < 64) {
    float s = 0.0f;
#pragma unroll
    for (int c = 0; c < DH; c += 4) {
      float4 z = *(const float4*)(Zs + tid * DH + c);
      s += z.x * z.x + z.y * z.y + z.z * z.z + z.w * z.w;
    }
    nrm[tid] = fmaxf(sqrtf(s), EPS);
  }
  __syncthreads();
  // normalized store (outer relu is identity on nonneg values)
#pragma unroll
  for (int j = 0; j < 8; ++j) {
    int e = j * 1024 + tid * 4;                     // 8192 floats total
    int r = e >> 7, c = e & 127;
    int grow = rowBase + r;
    if (grow < nrows) {
      float4 z = *(const float4*)(Zs + r * DH + c);
      float inv = 1.0f / nrm[r];
      float4 o{z.x * inv, z.y * inv, z.z * inv, z.w * inv};
      *(float4*)(OUT + (size_t)grow * DH + c) = o;
    }
  }
}

// ---- Kernel 5: eval edges: log_softmax(([x[s]|x[d]] @ W1 + b1) @ W2 + b2) ----
__global__ __launch_bounds__(256) void eval_kernel(
    const float* __restrict__ X1, const int* __restrict__ eidx,
    const float* __restrict__ W1, const float* __restrict__ b1,
    const float* __restrict__ W2, const float* __restrict__ b2,
    float* __restrict__ OUT, int nrows) {
  __shared__ __align__(32) _Float16 As[64 * 32];
  __shared__ __align__(32) _Float16 Bs[8 * 32 * 16];
  __shared__ float Hs[64 * DH];
  __shared__ int   sidx[64], didx[64];
  __shared__ float sW2[DH * 2];
  const int tid = threadIdx.x, lane = tid & 31, wave = tid >> 5;
  const int rowBase = blockIdx.x * 64;
  const int rt = wave & 3, ct0 = (wave >> 2) * 4;

  if (tid < 64) {
    int grow = rowBase + tid;
    sidx[tid] = (grow < nrows) ? eidx[grow] : 0;
    didx[tid] = (grow < nrows) ? eidx[nrows + grow] : 0;
  }
  sW2[tid] = W2[tid & (DH * 2 - 1)];
  __syncthreads();

  v8f acc[4];
#pragma unroll
  for (int t = 0; t < 4; ++t)
#pragma unroll
    for (int v = 0; v < 8; ++v) acc[t][v] = 0.0f;

  const int ar = tid >> 2, ac = (tid & 3) * 8;
  for (int kc = 0; kc < (2 * DH) / 32; ++kc) {
    {
      int grow = rowBase + ar;
      int col = kc * 32 + ac;
      float4 x0{0, 0, 0, 0}, x1{0, 0, 0, 0};
      if (grow < nrows) {
        const float* src = (col < DH)
            ? (X1 + (size_t)sidx[ar] * DH + col)
            : (X1 + (size_t)didx[ar] * DH + (col - DH));
        x0 = *(const float4*)src;
        x1 = *(const float4*)(src + 4);
      }
      cvt_store8(As + ar * 32 + ac, x0, x1);
    }
    stage_b(W1, kc, Bs, tid);
    __syncthreads();
    v16h a = frag_a(As, rt * 16, lane);
#pragma unroll
    for (int t = 0; t < 4; ++t)
      acc[t] = wmma_f16(a, frag_b(Bs, ct0 + t, lane), acc[t]);
    __syncthreads();
  }
#pragma unroll
  for (int t = 0; t < 4; ++t) {
    int n = (ct0 + t) * 16 + (lane & 15);
    float bv = b1[n];
#pragma unroll
    for (int v = 0; v < 8; ++v) {
      int m = rt * 16 + v + ((lane >> 4) << 3);
      Hs[m * DH + n] = acc[t][v] + bv;   // no relu before final linear
    }
  }
  __syncthreads();
  if (tid < 64) {
    int grow = rowBase + tid;
    if (grow < nrows) {
      float l0 = b2[0], l1 = b2[1];
#pragma unroll
      for (int k = 0; k < DH; k += 4) {
        float4 h = *(const float4*)(Hs + tid * DH + k);
        l0 += h.x * sW2[(k + 0) * 2] + h.y * sW2[(k + 1) * 2] +
              h.z * sW2[(k + 2) * 2] + h.w * sW2[(k + 3) * 2];
        l1 += h.x * sW2[(k + 0) * 2 + 1] + h.y * sW2[(k + 1) * 2 + 1] +
              h.z * sW2[(k + 2) * 2 + 1] + h.w * sW2[(k + 3) * 2 + 1];
      }
      float mx = fmaxf(l0, l1);
      float lse = mx + logf(expf(l0 - mx) + expf(l1 - mx));
      OUT[(size_t)grow * 2 + 0] = l0 - lse;
      OUT[(size_t)grow * 2 + 1] = l1 - lse;
    }
  }
}

// ---------------- host orchestration ----------------
extern "C" void kernel_launch(void* const* d_in, const int* in_sizes, int n_in,
                              void* d_out, int out_size, void* d_ws, size_t ws_size,
                              hipStream_t stream) {
  const float* x       = (const float*)d_in[0];
  const int*   eidx    = (const int*)d_in[1];
  const int*   evidx   = (const int*)d_in[2];
  const float* lin_w0  = (const float*)d_in[3];
  const float* lin_b0  = (const float*)d_in[4];
  const float* agg_w0  = (const float*)d_in[5];
  const float* agg_b0  = (const float*)d_in[6];
  const float* lin_w1  = (const float*)d_in[7];
  const float* lin_b1  = (const float*)d_in[8];
  const float* agg_w1  = (const float*)d_in[9];
  const float* agg_b1  = (const float*)d_in[10];
  const float* mp_w1   = (const float*)d_in[11];
  const float* mp_b1   = (const float*)d_in[12];
  const float* mp_w2   = (const float*)d_in[13];
  const float* mp_b2   = (const float*)d_in[14];
  float* out = (float*)d_out;

  const size_t nodeElems = (size_t)N_NODES * DH;
  float* h   = (float*)d_ws;          // [N,128]
  float* agg = h + nodeElems;         // [N,128]
  float* x1  = agg + nodeElems;       // [N,128]

  const int gemmBlocks = (N_NODES + 63) / 64;       // 782
  const int zeroBlocks = (int)((nodeElems / 4 + 255) / 256);
  const int edgeBlocks = (E_EDGES + 7) / 8;         // 100000
  const int evalBlocks = (N_EVAL + 63) / 64;        // 1563

  // layer 0
  lin_relu_kernel<<<gemmBlocks, 256, 0, stream>>>(x, lin_w0, lin_b0, h, N_NODES);
  zero_kernel<<<zeroBlocks, 256, 0, stream>>>((float4*)agg, (int)(nodeElems / 4));
  scatter_add_kernel<<<edgeBlocks, 256, 0, stream>>>(h, eidx, agg, E_EDGES);
  agg_norm_kernel<<<gemmBlocks, 256, 0, stream>>>(agg, x, agg_w0, agg_b0, x1, N_NODES);

  // layer 1 (concat-GEMM output written in place: each block reads only its
  // own 64 rows before writing them)
  lin_relu_kernel<<<gemmBlocks, 256, 0, stream>>>(x1, lin_w1, lin_b1, h, N_NODES);
  zero_kernel<<<zeroBlocks, 256, 0, stream>>>((float4*)agg, (int)(nodeElems / 4));
  scatter_add_kernel<<<edgeBlocks, 256, 0, stream>>>(h, eidx, agg, E_EDGES);
  agg_norm_kernel<<<gemmBlocks, 256, 0, stream>>>(agg, x1, agg_w1, agg_b1, x1, N_NODES);

  // link prediction head
  eval_kernel<<<evalBlocks, 256, 0, stream>>>(x1, evidx, mp_w1, mp_b1, mp_w2, mp_b2,
                                              out, N_EVAL);
}
